// MedicalMultiheadAttention_56135222559432
// MI455X (gfx1250) — matrix operand order repaired and checked by reading, not compiled, not run on hardware
//
#include <hip/hip_runtime.h>

// Problem constants (match reference).
#define BDIM 4
#define LDIM 2048
#define EDIM 1024
#define HNUM 16
#define HD   64

typedef __attribute__((ext_vector_type(16))) __bf16 v16bf;
typedef __attribute__((ext_vector_type(8)))  __bf16 v8bf;
typedef __attribute__((ext_vector_type(4)))  __bf16 v4bf;
typedef __attribute__((ext_vector_type(8)))  float  v8f;
typedef unsigned int v4u __attribute__((ext_vector_type(4)));
typedef int          v8i __attribute__((ext_vector_type(8)));
typedef int          v4i __attribute__((ext_vector_type(4)));

// ---- WMMA fragment helpers (CDNA5 16x16x32 bf16 layouts) ----
// A (16x32, LDS [M][K], K contiguous): lane half h reads 16 contiguous K
// values starting at kb+16h (two 16B LDS reads).
__device__ __forceinline__ v16bf frag_a(const __bf16* p) {
  union { v16bf v; v8bf h[2]; } u;
  u.h[0] = *(const v8bf*)(p);
  u.h[1] = *(const v8bf*)(p + 8);
  return u.v;
}
// B (32x16, LDS [N][K], K contiguous): lane half h reads 16 contiguous K
// values starting at kb+16h.
__device__ __forceinline__ v16bf frag_b(const __bf16* p) {
  union { v16bf v; v8bf h[2]; } u;
  u.h[0] = *(const v8bf*)(p);
  u.h[1] = *(const v8bf*)(p + 8);
  return u.v;
}
__device__ __forceinline__ v8f wmma_bf16(v16bf a, v16bf b, v8f c) {
  return __builtin_amdgcn_wmma_f32_16x16x32_bf16(false, a, false, b,
                                                 (short)0, c, false, false);
}

// ---- Tensor Data Mover path (guarded; fallback = synchronous copies) ----
#if defined(__has_builtin)
#if __has_builtin(__builtin_amdgcn_tensor_load_to_lds) && \
    __has_builtin(__builtin_amdgcn_s_wait_tensorcnt)
#define USE_TDM 1
#endif
#endif
#ifndef USE_TDM
#define USE_TDM 0
#endif

#if USE_TDM
__device__ __forceinline__ unsigned lds_offset(const void* p) {
  return (unsigned)(unsigned long long)
      (__attribute__((address_space(3))) const void*)p;
}
// 2D bf16 tile load: tile_x elems/row (must be mult of 8 here), tile_y rows,
// row stride in elements. LDS gets rows padded by 8 bf16 (pad_interval=32
// DWORDs -> code 4, pad_amount=4 DWORDs -> code 3) => 72-elem LDS stride.
__device__ __forceinline__ void tdm_load_tile_bf16(
    unsigned lds_off, const __bf16* gptr,
    unsigned tile_x, unsigned tile_y, unsigned row_stride_elems)
{
  const unsigned long long ga = (unsigned long long)gptr;
  v4u g0;
  g0[0] = 1u;                                            // count=1 (valid D#)
  g0[1] = lds_off;                                       // LDS byte address
  g0[2] = (unsigned)(ga & 0xFFFFFFFFu);                  // global_addr lo
  g0[3] = (unsigned)((ga >> 32) & 0x01FFFFFFu) | (2u << 30); // addr hi | type=2
  const unsigned td0 = 1u << 20, td1 = 1u << 20;         // huge dims: no OOB clip
  v8i g1;
  g1[0] = (int)((1u << 16)        // data_size = 1 -> 2 bytes
              | (1u << 20)        // pad_enable
              | (4u << 22)        // pad_interval: 32 DWORDs (one 64-elem row)
              | (3u << 25));      // pad_amount: 4 DWORDs (8 bf16)
  g1[1] = (int)((td0 & 0xFFFFu) << 16);                  // tensor_dim0 lo
  g1[2] = (int)((td0 >> 16) | ((td1 & 0xFFFFu) << 16));  // td0 hi | td1 lo
  g1[3] = (int)((td1 >> 16) | (tile_x << 16));           // td1 hi | tile_dim0
  g1[4] = (int)tile_y;                                   // tile_dim1 (dim2=0)
  g1[5] = (int)row_stride_elems;                         // tensor_dim0_stride lo
  g1[6] = 0;                                             // stride hi | dim1_stride
  g1[7] = 0;
  v4i z4; z4[0] = 0; z4[1] = 0; z4[2] = 0; z4[3] = 0;    // groups 2/3 unused (2D)
  v8i z8; z8[0] = 0; z8[1] = 0; z8[2] = 0; z8[3] = 0;
          z8[4] = 0; z8[5] = 0; z8[6] = 0; z8[7] = 0;
  // clang-23 / amdgpu-toolchain arity: 6 args (extra int32x8 group + cpol).
  __builtin_amdgcn_tensor_load_to_lds(g0, g1, z4, z4, z8, 0);
}
#endif

// =====================================================================
// Kernel 1: QKV projection. z=0/1/2 -> q/k/v. GEMM M=8192,N=1024,K=1024.
// out[m,n] = sum_k X[m,k] * W[n,k]. Writes bf16 workspace:
//   q,k: (b,h,l,d) row-major.  v: (b,h,d,l)  (pre-transposed for PV WMMA).
// =====================================================================
__global__ __launch_bounds__(256) void qkv_proj_kernel(
    const float* __restrict__ query, const float* __restrict__ key,
    const float* __restrict__ value, const float* __restrict__ wqkv,
    __bf16* __restrict__ qws, __bf16* __restrict__ kws, __bf16* __restrict__ vws)
{
  __shared__ __bf16 As[128 * 72];   // [M=128][K=64] pad->72
  __shared__ __bf16 Bs[128 * 72];   // [N=128][K=64] pad->72
  const int tid  = threadIdx.x;
  const int lane = tid & 31, w = tid >> 5;
  const int half = lane >> 4, lr = lane & 15;
  const int m0 = blockIdx.y * 128, n0 = blockIdx.x * 128;
  const int z  = blockIdx.z;
  const float* src = (z == 0) ? query : (z == 1) ? key : value;
  const float* W   = wqkv + (size_t)z * EDIM * EDIM;
  const int wm = (w & 3) * 32, wn = (w >> 2) * 64;

  v8f acc[2][4] = {};

  for (int kt = 0; kt < EDIM / 64; ++kt) {
    // Stage 128x64 tiles of X and W, fusing f32->bf16. 2048 float4 each.
#pragma unroll
    for (int i = 0; i < 8; ++i) {
      int id = tid + i * 256;
      int row = id >> 4, c4 = id & 15;
      float4 f = *(const float4*)(src + (size_t)(m0 + row) * EDIM + kt * 64 + c4 * 4);
      v4bf a4; a4[0] = (__bf16)f.x; a4[1] = (__bf16)f.y;
               a4[2] = (__bf16)f.z; a4[3] = (__bf16)f.w;
      *(v4bf*)(&As[row * 72 + c4 * 4]) = a4;
      float4 g = *(const float4*)(W + (size_t)(n0 + row) * EDIM + kt * 64 + c4 * 4);
      v4bf b4; b4[0] = (__bf16)g.x; b4[1] = (__bf16)g.y;
               b4[2] = (__bf16)g.z; b4[3] = (__bf16)g.w;
      *(v4bf*)(&Bs[row * 72 + c4 * 4]) = b4;
    }
    __syncthreads();

#pragma unroll
    for (int ks = 0; ks < 64; ks += 32) {
      v16bf afv[2], bfv[4];
#pragma unroll
      for (int rt = 0; rt < 2; ++rt)
        afv[rt] = frag_a(&As[(wm + rt * 16 + lr) * 72 + ks + half * 16]);
#pragma unroll
      for (int j = 0; j < 4; ++j)
        bfv[j] = frag_b(&Bs[(wn + j * 16 + lr) * 72 + ks + half * 16]);
#pragma unroll
      for (int rt = 0; rt < 2; ++rt)
#pragma unroll
        for (int j = 0; j < 4; ++j)
          acc[rt][j] = wmma_bf16(afv[rt], bfv[j], acc[rt][j]);
    }
    __syncthreads();
  }

  // Epilogue. b is constant per block (128 | 2048). C layout: row = r+8*half.
  const int bidx  = m0 >> 11;
  const int lwave = (m0 & (LDIM - 1)) + wm + half * 8;
  if (z == 2) {
    // vws[((b*H+h)*HD+d)*L + l]: 8 rows per lane are 8 consecutive l -> b128.
#pragma unroll
    for (int rt = 0; rt < 2; ++rt) {
      const int lbase = lwave + rt * 16;
#pragma unroll
      for (int j = 0; j < 4; ++j) {
        const int n = n0 + wn + j * 16 + lr;
        const int h = n >> 6, d = n & 63;
        v8bf pk;
#pragma unroll
        for (int r = 0; r < 8; ++r) pk[r] = (__bf16)acc[rt][j][r];
        *(v8bf*)(vws + (((size_t)bidx * HNUM + h) * HD + d) * LDIM + lbase) = pk;
      }
    }
  } else {
    __bf16* dst = (z == 0) ? qws : kws;
    const float sc = (z == 0) ? 0.125f : 1.0f;   // HD^-0.5 folded into q
#pragma unroll
    for (int rt = 0; rt < 2; ++rt) {
      const int lbase = lwave + rt * 16;
#pragma unroll
      for (int j = 0; j < 4; ++j) {
        const int n = n0 + wn + j * 16 + lr;
        const int h = n >> 6, d = n & 63;
        __bf16* p = dst + (((size_t)bidx * HNUM + h) * LDIM + lbase) * HD + d;
#pragma unroll
        for (int r = 0; r < 8; ++r)
          p[(size_t)r * HD] = (__bf16)(acc[rt][j][r] * sc);
      }
    }
  }
}

// =====================================================================
// Kernel 2: Flash attention per (b,h). 128 query rows/block, 8 waves, each
// wave owns 16 rows. KV streamed in blocks of 64 via the TDM when available.
// =====================================================================
__global__ __launch_bounds__(256) void flash_attn_kernel(
    const __bf16* __restrict__ qws, const __bf16* __restrict__ kws,
    const __bf16* __restrict__ vws, __bf16* __restrict__ aout)
{
  __shared__ __bf16 Qs[128 * 72];       // [qrow][d]
  __shared__ __bf16 Ks[64 * 72];        // [kvrow][d]  (K^T in [N][K] form)
  __shared__ __bf16 Vt[64 * 72];        // [d][kv]     (V in [N][K] form)
  __shared__ __bf16 Ps[8 * 16 * 72];    // wave-private P strips [qrow][kv]
  const int tid  = threadIdx.x;
  const int lane = tid & 31, w = tid >> 5;
  const int half = lane >> 4, lr = lane & 15;
  const int bh = blockIdx.y;
  const int b = bh >> 4, h = bh & 15;
  const int q0 = blockIdx.x * 128;
  const size_t qkbase = (size_t)bh * LDIM * HD;
  const size_t vbase  = (size_t)bh * HD * LDIM;

  // Stage Q block once (visibility covered by first in-loop barrier).
#if USE_TDM
  if (w == 0)
    tdm_load_tile_bf16(lds_offset(Qs), qws + qkbase + (size_t)q0 * HD, 64, 128, 64);
#else
#pragma unroll
  for (int i = 0; i < 4; ++i) {
    int id = tid + i * 256;
    int row = id >> 3, c8 = id & 7;
    v8bf v = *(const v8bf*)(qws + qkbase + (size_t)(q0 + row) * HD + c8 * 8);
    *(v8bf*)(&Qs[row * 72 + c8 * 8]) = v;
  }
#endif

  float mrow[8], lrow[8];
#pragma unroll
  for (int r = 0; r < 8; ++r) { mrow[r] = -1e30f; lrow[r] = 0.0f; }
  v8f o[4] = {};
  const int qm = w * 16;
  __bf16* myP = &Ps[w * 16 * 72];

  for (int kv0 = 0; kv0 < LDIM; kv0 += 64) {
    __syncthreads();   // previous iteration's readers done
#if USE_TDM
    if (w == 0) {
      tdm_load_tile_bf16(lds_offset(Ks), kws + qkbase + (size_t)kv0 * HD, 64, 64, 64);
      tdm_load_tile_bf16(lds_offset(Vt), vws + vbase + kv0, 64, 64, LDIM);
      __builtin_amdgcn_s_wait_tensorcnt(0);   // TENSORcnt is per-wave
    }
#else
#pragma unroll
    for (int i = 0; i < 2; ++i) {
      int id = tid + i * 256;
      int row = id >> 3, c8 = id & 7;
      v8bf kv = *(const v8bf*)(kws + qkbase + (size_t)(kv0 + row) * HD + c8 * 8);
      *(v8bf*)(&Ks[row * 72 + c8 * 8]) = kv;
      v8bf vv = *(const v8bf*)(vws + vbase + (size_t)row * LDIM + kv0 + c8 * 8);
      *(v8bf*)(&Vt[row * 72 + c8 * 8]) = vv;
    }
#endif
    __syncthreads();

    // S = Q*K^T (16 rows x 64 kv, K=64 via two x32 steps)
    v8f s[4] = {};
#pragma unroll
    for (int ks = 0; ks < 64; ks += 32) {
      v16bf qa = frag_a(&Qs[(qm + lr) * 72 + ks + half * 16]);
#pragma unroll
      for (int t = 0; t < 4; ++t) {
        v16bf kb = frag_b(&Ks[(t * 16 + lr) * 72 + ks + half * 16]);
        s[t] = wmma_bf16(qa, kb, s[t]);
      }
    }

    // Online softmax: row r of this half lives across its 16 lanes.
#pragma unroll
    for (int r = 0; r < 8; ++r) {
      float mx = fmaxf(fmaxf(s[0][r], s[1][r]), fmaxf(s[2][r], s[3][r]));
#pragma unroll
      for (int msk = 1; msk < 16; msk <<= 1)
        mx = fmaxf(mx, __shfl_xor(mx, msk, 32));
      float mnew = fmaxf(mrow[r], mx);
      float corr = __expf(mrow[r] - mnew);
      mrow[r] = mnew;
      float rs = 0.0f;
#pragma unroll
      for (int t = 0; t < 4; ++t) {
        float p = __expf(s[t][r] - mnew);
        s[t][r] = p;
        rs += p;
      }
#pragma unroll
      for (int msk = 1; msk < 16; msk <<= 1)
        rs += __shfl_xor(rs, msk, 32);
      lrow[r] = lrow[r] * corr + rs;
#pragma unroll
      for (int j = 0; j < 4; ++j) o[j][r] *= corr;
    }

    // Write P (wave-private; LDS ops are in-order within a wave).
#pragma unroll
    for (int t = 0; t < 4; ++t)
#pragma unroll
      for (int r = 0; r < 8; ++r)
        myP[(r + half * 8) * 72 + t * 16 + lr] = (__bf16)s[t][r];

    // O += P*V  (A = P [16x64], B = V as [d][kv])
#pragma unroll
    for (int ks = 0; ks < 64; ks += 32) {
      v16bf pa = frag_a(&myP[lr * 72 + ks + half * 16]);
#pragma unroll
      for (int j = 0; j < 4; ++j) {
        v16bf vb = frag_b(&Vt[(j * 16 + lr) * 72 + ks + half * 16]);
        o[j] = wmma_bf16(pa, vb, o[j]);
      }
    }
  }

  // Normalize + write (B,L,E) bf16; reshape(B,L,H,HD)==(B,L,E).
  __bf16* obase = aout + ((size_t)b * LDIM + q0 + qm + half * 8) * EDIM
                + h * HD + lr;
#pragma unroll
  for (int r = 0; r < 8; ++r) {
    const float inv = 1.0f / lrow[r];
#pragma unroll
    for (int j = 0; j < 4; ++j)
      obase[(size_t)r * EDIM + j * 16] = (__bf16)(o[j][r] * inv);
  }
}

// =====================================================================
// Kernel 3: Output projection + bias, f32 out. M=8192,N=1024,K=1024.
// =====================================================================
__global__ __launch_bounds__(256) void out_proj_kernel(
    const __bf16* __restrict__ ain, const float* __restrict__ wout,
    const float* __restrict__ bias, float* __restrict__ out)
{
  __shared__ __bf16 As[128 * 72];
  __shared__ __bf16 Bs[128 * 72];
  const int tid  = threadIdx.x;
  const int lane = tid & 31, w = tid >> 5;
  const int half = lane >> 4, lr = lane & 15;
  const int m0 = blockIdx.y * 128, n0 = blockIdx.x * 128;
  const int wm = (w & 3) * 32, wn = (w >> 2) * 64;
  v8f acc[2][4] = {};

  for (int kt = 0; kt < EDIM / 64; ++kt) {
#pragma unroll
    for (int i = 0; i < 4; ++i) {        // A: 1024 chunks of 8 bf16
      int id = tid + i * 256;
      int row = id >> 3, c8 = id & 7;
      v8bf av = *(const v8bf*)(ain + (size_t)(m0 + row) * EDIM + kt * 64 + c8 * 8);
      *(v8bf*)(&As[row * 72 + c8 * 8]) = av;
    }
#pragma unroll
    for (int i = 0; i < 8; ++i) {        // B: 2048 float4 of W, convert
      int id = tid + i * 256;
      int row = id >> 4, c4 = id & 15;
      float4 g = *(const float4*)(wout + (size_t)(n0 + row) * EDIM + kt * 64 + c4 * 4);
      v4bf b4; b4[0] = (__bf16)g.x; b4[1] = (__bf16)g.y;
               b4[2] = (__bf16)g.z; b4[3] = (__bf16)g.w;
      *(v4bf*)(&Bs[row * 72 + c4 * 4]) = b4;
    }
    __syncthreads();

#pragma unroll
    for (int ks = 0; ks < 64; ks += 32) {
      v16bf afv[2], bfv[4];
#pragma unroll
      for (int rt = 0; rt < 2; ++rt)
        afv[rt] = frag_a(&As[(wm + rt * 16 + lr) * 72 + ks + half * 16]);
#pragma unroll
      for (int j = 0; j < 4; ++j)
        bfv[j] = frag_b(&Bs[(wn + j * 16 + lr) * 72 + ks + half * 16]);
#pragma unroll
      for (int rt = 0; rt < 2; ++rt)
#pragma unroll
        for (int j = 0; j < 4; ++j)
          acc[rt][j] = wmma_bf16(afv[rt], bfv[j], acc[rt][j]);
    }
    __syncthreads();
  }

#pragma unroll
  for (int j = 0; j < 4; ++j) {
    const int n = n0 + wn + j * 16 + lr;
    const float bj = bias[n];
    float* pj = out + (size_t)(m0 + wm + half * 8) * EDIM + n;
#pragma unroll
    for (int rt = 0; rt < 2; ++rt)
#pragma unroll
      for (int r = 0; r < 8; ++r)
        __builtin_nontemporal_store(acc[rt][j][r] + bj,
                                    pj + (size_t)(rt * 16 + r) * EDIM);
  }
}

// =====================================================================
extern "C" void kernel_launch(void* const* d_in, const int* in_sizes, int n_in,
                              void* d_out, int out_size, void* d_ws, size_t ws_size,
                              hipStream_t stream) {
  (void)in_sizes; (void)n_in; (void)out_size; (void)ws_size;
  const float* query = (const float*)d_in[0];
  const float* key   = (const float*)d_in[1];
  const float* value = (const float*)d_in[2];
  const float* wqkv  = (const float*)d_in[3];
  const float* wout  = (const float*)d_in[4];
  const float* bout  = (const float*)d_in[5];
  float* out = (float*)d_out;

  const size_t nelem = (size_t)BDIM * HNUM * LDIM * HD;  // 8.39M per buffer
  __bf16* qws = (__bf16*)d_ws;
  __bf16* kws = qws + nelem;
  __bf16* vws = kws + nelem;
  __bf16* aws = vws + nelem;   // attention output (B, L, E) bf16

  qkv_proj_kernel<<<dim3(EDIM / 128, (BDIM * LDIM) / 128, 3), 256, 0, stream>>>(
      query, key, value, wqkv, qws, kws, vws);
  flash_attn_kernel<<<dim3(LDIM / 128, BDIM * HNUM), 256, 0, stream>>>(
      qws, kws, vws, aws);
  out_proj_kernel<<<dim3(EDIM / 128, (BDIM * LDIM) / 128), 256, 0, stream>>>(
      aws, wout, bout, out);
}
